// LoRAConnector_11355893530725
// MI455X (gfx1250) — compile-verified, matrix operand order checked by reading, not run on hardware
//
#include <hip/hip_runtime.h>
#include <hip/hip_bf16.h>

// ---- shapes (fixed by setup_inputs) ----
#define NA   16      // adapters
#define NS   4       // streams n
#define CSZ  1024    // C
#define NC   4096    // n*C
#define TSEQ 2048    // T
#define NB   8       // B
#define NTOK 16384   // B*T
#define KCH  128     // 4096 / 32  (K chunks per WMMA-K of 32)
#define EPSV 1e-5f
#define SKIT 20

typedef __bf16 v16bf __attribute__((ext_vector_type(16)));
typedef float  v8f   __attribute__((ext_vector_type(8)));
typedef float  v4f   __attribute__((ext_vector_type(4)));   // native vector for NT builtins

// ============================================================================
// Kernel 1: pack B operand.
// packedB[a][kc][nh][lane][j]  (bf16, 16 elems = 32B per (a,kc,nh,lane))
// Element (k,col) with col = nh*16 + (lane&15),
//   k = kc*32 + (j<8 ? 8*h + j : 16 + 8*h + (j-8)),  h = lane>>4
// (mirrors the 16-bit WMMA A/B fragment striping).
// Value = input_norm_weight[a][k] * alpha_sel[a] * phi_sel[a][k][col'] ; cols
// 0..3 = phi_pre, 4..7 = phi_post, 8..23 = phi_res, 24..31 = zero padding.
// ============================================================================
__global__ __launch_bounds__(256) void pack_b_kernel(
    const float* __restrict__ w,
    const float* __restrict__ phi_pre,  const float* __restrict__ phi_post,
    const float* __restrict__ phi_res,
    const float* __restrict__ alpha_pre, const float* __restrict__ alpha_post,
    const float* __restrict__ alpha_res,
    __bf16* __restrict__ packedB)
{
    int tid  = blockIdx.x * blockDim.x + threadIdx.x;   // 16*128*2*32 = 131072
    int lane = tid & 31;
    int kc   = (tid >> 6) & (KCH - 1);
    int a    = tid >> 13;
    int n    = lane & 15;
    int h    = lane >> 4;
    int col  = ((tid >> 5) & 1) * 16 + n;

    v16bf frag;
#pragma unroll
    for (int j = 0; j < 16; ++j) {
        int k = kc * 32 + (j < 8 ? 8 * h + j : 16 + 8 * h + (j - 8));
        float v = 0.0f;
        if (col < 4)
            v = alpha_pre[a]  * phi_pre [(size_t)(a * NC + k) * NS + col]       * w[a * NC + k];
        else if (col < 8)
            v = alpha_post[a] * phi_post[(size_t)(a * NC + k) * NS + (col - 4)] * w[a * NC + k];
        else if (col < 24)
            v = alpha_res[a]  * phi_res [(size_t)(a * NC + k) * 16 + (col - 8)] * w[a * NC + k];
        frag[j] = (__bf16)v;
    }
    // fragment index == tid by construction
    *(((v16bf*)packedB) + tid) = frag;
}

// stabilized logsumexp of 4 (matches jax.nn.logsumexp)
__device__ __forceinline__ float lse4(float a, float b, float c, float d) {
    float mx = fmaxf(fmaxf(a, b), fmaxf(c, d));
    return mx + __logf(__expf(a - mx) + __expf(b - mx) + __expf(c - mx) + __expf(d - mx));
}
__device__ __forceinline__ float sigm(float x) { return 1.0f / (1.0f + __expf(-x)); }

// ============================================================================
// Kernel 2: fused main kernel. One wave = one 16-token tile.
//  Phase 1: K-loop (unrolled x2), v_wmma_f32_16x16x32_bf16 ×2 per chunk
//           (N=0..15, 16..31); the same f32 loads feed a float4 sum-of-squares
//           accumulator (4 independent FMA chains, no serial add chain).
//  Phase 2: transpose D via LDS; lanes 0..15 each run one token's gates +
//           20-iter log-space Sinkhorn in registers; emit the fused 4×4 mix
//           W = M + h_post ⊗ h_pre to LDS.
//  Phase 3: streaming output out[tok] = W @ x[tok] with v4f FMAs.
//           Re-read of x is last-use -> non-temporal loads; out store is
//           streaming write-once -> non-temporal stores (keep L2 for the
//           phase-1 -> phase-3 x reuse of the *next* tiles).
// ============================================================================
__global__ __launch_bounds__(256) void lora_main_kernel(
    const float* __restrict__ x,
    const int*   __restrict__ adapter_indices,
    const float* __restrict__ b_pre, const float* __restrict__ b_post,
    const float* __restrict__ b_res,
    const __bf16* __restrict__ packedB,
    float* __restrict__ out)
{
    __shared__ float dmat[8][16][32];   // [wave][token][col]
    __shared__ float wmat[8][16][16];   // [wave][token][4x4 mix]

    const int wid   = threadIdx.x >> 5;
    const int lane  = threadIdx.x & 31;
    const int tile  = blockIdx.x * 8 + wid;          // 1024 tiles
    const int tok0  = tile * 16;
    const int bidx  = tok0 >> 11;                    // / T (2048)
    const int a     = adapter_indices[bidx];
    const int m     = lane & 15;                     // token-in-tile / column
    const int h     = lane >> 4;
    const int r0    = 8 * h;
    const int r1    = 16 + 8 * h;

    const float* xrow = x + (size_t)(tok0 + m) * NC;
    const v16bf* bfr  = (const v16bf*)packedB + (size_t)a * KCH * 2 * 32;

    v8f acc0 = {};   // D columns 0..15
    v8f acc1 = {};   // D columns 16..31 (24..31 are zero-padded B)
    v4f ssq4 = {0.f, 0.f, 0.f, 0.f};                 // 4 independent chains

#pragma unroll 2
    for (int kc = 0; kc < KCH; ++kc) {
        const int kb = kc * 32;
        v4f f0 = *(const v4f*)(xrow + kb + r0);
        v4f f1 = *(const v4f*)(xrow + kb + r0 + 4);
        v4f f2 = *(const v4f*)(xrow + kb + r1);
        v4f f3 = *(const v4f*)(xrow + kb + r1 + 4);

        ssq4.x = fmaf(f0.x, f0.x, fmaf(f1.x, f1.x, fmaf(f2.x, f2.x, fmaf(f3.x, f3.x, ssq4.x))));
        ssq4.y = fmaf(f0.y, f0.y, fmaf(f1.y, f1.y, fmaf(f2.y, f2.y, fmaf(f3.y, f3.y, ssq4.y))));
        ssq4.z = fmaf(f0.z, f0.z, fmaf(f1.z, f1.z, fmaf(f2.z, f2.z, fmaf(f3.z, f3.z, ssq4.z))));
        ssq4.w = fmaf(f0.w, f0.w, fmaf(f1.w, f1.w, fmaf(f2.w, f2.w, fmaf(f3.w, f3.w, ssq4.w))));

        v16bf af;
        af[0] =(__bf16)f0.x; af[1] =(__bf16)f0.y; af[2] =(__bf16)f0.z; af[3] =(__bf16)f0.w;
        af[4] =(__bf16)f1.x; af[5] =(__bf16)f1.y; af[6] =(__bf16)f1.z; af[7] =(__bf16)f1.w;
        af[8] =(__bf16)f2.x; af[9] =(__bf16)f2.y; af[10]=(__bf16)f2.z; af[11]=(__bf16)f2.w;
        af[12]=(__bf16)f3.x; af[13]=(__bf16)f3.y; af[14]=(__bf16)f3.z; af[15]=(__bf16)f3.w;

        v16bf bf0 = bfr[kc * 64 + lane];        // nh = 0
        v16bf bf1 = bfr[kc * 64 + 32 + lane];   // nh = 1

        acc0 = __builtin_amdgcn_wmma_f32_16x16x32_bf16(
                   false, af, false, bf0, (short)0, acc0, false, false);
        acc1 = __builtin_amdgcn_wmma_f32_16x16x32_bf16(
                   false, af, false, bf1, (short)0, acc1, false, false);
    }

    float sumsq = (ssq4.x + ssq4.y) + (ssq4.z + ssq4.w);
    // full per-token sum of squares (lanes l and l^16 cover disjoint K halves)
    sumsq += __shfl_xor(sumsq, 16, 32);

    // D layout: vgpr v, lane l -> row M = v + 8*(l>>4), col N = l&15
#pragma unroll
    for (int v = 0; v < 8; ++v) {
        dmat[wid][v + 8 * h][m]      = acc0[v];
        dmat[wid][v + 8 * h][16 + m] = acc1[v];
    }
    __syncthreads();

    if (lane < 16) {
        const float inv_rms = rsqrtf(sumsq * (1.0f / NC) + EPSV);
        const float* drow = dmat[wid][lane];

        float hpre[4], hpost[4], r[16];
#pragma unroll
        for (int j = 0; j < 4; ++j) {
            hpre[j]  =        sigm(inv_rms * drow[j]     + b_pre [a * 4 + j]);
            hpost[j] = 2.0f * sigm(inv_rms * drow[4 + j] + b_post[a * 4 + j]);
        }
#pragma unroll
        for (int i = 0; i < 16; ++i)
            r[i] = inv_rms * drow[8 + i] + b_res[a * 16 + i];

        // log-space Sinkhorn-Knopp, 20 iterations
#pragma unroll 1
        for (int it = 0; it < SKIT; ++it) {
#pragma unroll
            for (int i = 0; i < 4; ++i) {
                float l = lse4(r[i*4+0], r[i*4+1], r[i*4+2], r[i*4+3]);
                r[i*4+0] -= l; r[i*4+1] -= l; r[i*4+2] -= l; r[i*4+3] -= l;
            }
#pragma unroll
            for (int j = 0; j < 4; ++j) {
                float l = lse4(r[j], r[4+j], r[8+j], r[12+j]);
                r[j] -= l; r[4+j] -= l; r[8+j] -= l; r[12+j] -= l;
            }
        }
        // fused mix: W[k][j] = M[k][j] + h_post[k]*h_pre[j]
#pragma unroll
        for (int k = 0; k < 4; ++k)
#pragma unroll
            for (int j = 0; j < 4; ++j)
                wmat[wid][lane][k * 4 + j] = __expf(r[k * 4 + j]) + hpost[k] * hpre[j];
    }
    __syncthreads();

    // Phase 3: out[tok] = W @ x[tok]  (4x4 mixing of the 4 streams)
    v4f W0 = *(const v4f*)&wmat[wid][m][0];
    v4f W1 = *(const v4f*)&wmat[wid][m][4];
    v4f W2 = *(const v4f*)&wmat[wid][m][8];
    v4f W3 = *(const v4f*)&wmat[wid][m][12];

    const float* xr   = x   + (size_t)(tok0 + m) * NC;
    float*       orow = out + (size_t)(tok0 + m) * NC;
    const int cbase = h * 512;     // lane halves split the C range

    for (int cb = 0; cb < 128; ++cb) {
        const int c = cbase + cb * 4;
        v4f x0 = __builtin_nontemporal_load((const v4f*)(xr + 0 * CSZ + c));
        v4f x1 = __builtin_nontemporal_load((const v4f*)(xr + 1 * CSZ + c));
        v4f x2 = __builtin_nontemporal_load((const v4f*)(xr + 2 * CSZ + c));
        v4f x3 = __builtin_nontemporal_load((const v4f*)(xr + 3 * CSZ + c));
#pragma unroll
        for (int k = 0; k < 4; ++k) {
            v4f Wk = (k == 0) ? W0 : (k == 1) ? W1 : (k == 2) ? W2 : W3;
            v4f o;
            o.x = Wk.x * x0.x + Wk.y * x1.x + Wk.z * x2.x + Wk.w * x3.x;
            o.y = Wk.x * x0.y + Wk.y * x1.y + Wk.z * x2.y + Wk.w * x3.y;
            o.z = Wk.x * x0.z + Wk.y * x1.z + Wk.z * x2.z + Wk.w * x3.z;
            o.w = Wk.x * x0.w + Wk.y * x1.w + Wk.z * x2.w + Wk.w * x3.w;
            __builtin_nontemporal_store(o, (v4f*)(orow + k * CSZ + c));
        }
    }
}

extern "C" void kernel_launch(void* const* d_in, const int* in_sizes, int n_in,
                              void* d_out, int out_size, void* d_ws, size_t ws_size,
                              hipStream_t stream) {
    const float* x      = (const float*)d_in[0];
    const int*   aidx   = (const int*)  d_in[1];
    const float* w      = (const float*)d_in[2];
    const float* ppre   = (const float*)d_in[3];
    const float* ppost  = (const float*)d_in[4];
    const float* pres   = (const float*)d_in[5];
    const float* bpre   = (const float*)d_in[6];
    const float* bpost  = (const float*)d_in[7];
    const float* bres   = (const float*)d_in[8];
    const float* apre   = (const float*)d_in[9];
    const float* apost  = (const float*)d_in[10];
    const float* ares   = (const float*)d_in[11];
    float*   out    = (float*)d_out;
    __bf16*  packB  = (__bf16*)d_ws;   // 16*128*2*32*16 bf16 = 4 MB

    // 131072 fragment threads
    pack_b_kernel<<<512, 256, 0, stream>>>(w, ppre, ppost, pres, apre, apost, ares, packB);
    // 1024 tiles of 16 tokens, 8 waves per block
    lora_main_kernel<<<128, 256, 0, stream>>>(x, aidx, bpre, bpost, bres, packB, out);
}
